// QuantizedMoeExpertsBase_17867063951961
// MI455X (gfx1250) — matrix operand order repaired and checked by reading, not compiled, not run on hardware
//
#include <hip/hip_runtime.h>
#include <hip/hip_bf16.h>
#include <cstdint>
#include <cstddef>

// ---------------- problem constants ----------------
constexpr int T = 2048, TOPK = 6, E = 32, D = 2048, F = 1408, S = 2;
constexpr int C  = T * TOPK / E;   // 384 rows per routed expert (balanced)
constexpr int TK = T * TOPK;       // 12288 token-expert pairs
constexpr int F2 = 2 * F;          // 2816 (gate+up merged)
constexpr int ROUTED_ROWS = E * C; // 12288

// ---------------- workspace layout (bytes) ----------------
constexpr size_t OFF_TOK   = 0;                                   // int[TK]
constexpr size_t OFF_WPAIR = OFF_TOK   + sizeof(int)   * TK;      // float[TK]
constexpr size_t OFF_POS   = OFF_WPAIR + sizeof(float) * TK;      // int[TK]
constexpr size_t OFF_XH    = 147456;                              // f16[T*D] (256B aligned)
constexpr size_t OFF_ACT   = OFF_XH  + sizeof(_Float16) * (size_t)T * D;
constexpr size_t ACT_SHARED_OFF_H = (size_t)ROUTED_ROWS * F;      // halves, start of shared act [T, 2F]
constexpr size_t OFF_DOWN  = OFF_ACT +
    sizeof(_Float16) * ((size_t)ROUTED_ROWS * F + (size_t)T * F2); // f16[E*C, D]

typedef _Float16 v16h __attribute__((ext_vector_type(16)));
typedef float    v8f  __attribute__((ext_vector_type(8)));
typedef _Float16 h2   __attribute__((ext_vector_type(2)));
typedef _Float16 h4   __attribute__((ext_vector_type(4)));
typedef float    f4   __attribute__((ext_vector_type(4)));

// async global->LDS copy of one dword per lane (gather + LDS scatter in one op).
// Tracked on ASYNCcnt; wave must s_wait_asynccnt 0 before the consuming barrier.
__device__ __forceinline__ void async_copy_b32(uint32_t lds_byte_addr, const void* gptr) {
    asm volatile("global_load_async_to_lds_b32 %0, %1, off"
                 :: "v"(lds_byte_addr), "v"((uint64_t)(uintptr_t)gptr)
                 : "memory");
}
__device__ __forceinline__ void wait_async0() {
    asm volatile("s_wait_asynccnt 0" ::: "memory");
}

// ============================================================
// 1) dispatch: stable sort-by-expert via per-expert wave32 scan
//    (exactly reproduces argsort(flat_e, stable=True))
// ============================================================
__global__ __launch_bounds__(32)
void dispatch_kernel(const int* __restrict__ idx, const float* __restrict__ wts,
                     int* __restrict__ tok, float* __restrict__ wp, int* __restrict__ pos)
{
    const int e = blockIdx.x;
    const int lane = threadIdx.x;        // wave32
    const unsigned lt_mask = (1u << lane) - 1u;
    int base = 0;
    for (int i0 = 0; i0 < TK; i0 += 32) {
        const int p = i0 + lane;
        const bool m = (idx[p] == e);
        const unsigned mask = (unsigned)__ballot(m);   // low 32 bits valid on wave32
        if (m) {
            const int slot = e * C + base + __builtin_popcount(mask & lt_mask);
            tok[slot] = p / TOPK;
            wp[slot]  = wts[p];
            pos[p]    = slot;
        }
        base += __builtin_popcount(mask);
    }
}

// ============================================================
// 2) convert x (fp32) -> f16 once; reused by routed gather + shared
// ============================================================
__global__ __launch_bounds__(256)
void convert_x_kernel(const float* __restrict__ x, _Float16* __restrict__ xh)
{
    const int i = blockIdx.x * 256 + threadIdx.x;    // i < T*D/4
    f4 v = ((const f4*)x)[i];
    h4 h;
    h[0] = (_Float16)v[0]; h[1] = (_Float16)v[1];
    h[2] = (_Float16)v[2]; h[3] = (_Float16)v[3];
    ((h4*)xh)[i] = h;
}

// ---- WMMA fragment-major LDS addressing helpers (ISA §7.12.2) ----
// A (16x32 f16): lane L = (m&15)+16*h holds K in {0..7,16..23} (h=0) / {8..15,24..31} (h=1)
// B (32x16 f16): lane L = n + 16*h holds K in {0..7,16..23} (h=0) / {8..15,24..31} (h=1)
__device__ __forceinline__ int a_dword_idx(int mr, int j /* k-pair 0..15 */) {
    const int h = (j >> 2) & 1;
    const int v = ((j & 8) ? 4 : 0) + (j & 3);
    return (mr >> 4) * 256 + ((mr & 15) + 16 * h) * 8 + v;
}
__device__ __forceinline__ int b_dword_idx(int k /* even, 0..30 */, int n /*0..63*/) {
    // halves (k, k+1) share one dword in the fragment-major layout
    return (n >> 4) * 256 + ((n & 15) + 16 * ((k >> 3) & 1)) * 8 + (k >> 4) * 4 + ((k & 7) >> 1);
}

// ============================================================
// 3) fused gate_up grouped GEMM (routed + shared) -> silu*mul -> act (f16)
//    block tile 128x64, 8 waves (4Mx2N), wave tile 32x32, 2x2 frags x {g,u}
// ============================================================
__global__ __launch_bounds__(256)
void gateup_kernel(const _Float16* __restrict__ xh,
                   const float* __restrict__ wgu,     // [E, D, 2F]
                   const float* __restrict__ wgu_sh,  // [S, D, 2F]
                   const int*  __restrict__ tok_sorted,
                   _Float16*   __restrict__ act)
{
    __shared__ _Float16 sA [128 * 32];   // 8 KB, fragment-major
    __shared__ _Float16 sBg[ 32 * 64];   // 4 KB
    __shared__ _Float16 sBu[ 32 * 64];   // 4 KB
    __shared__ int stok[128];

    const int tid = threadIdx.x, lane = tid & 31, wid = tid >> 5;
    const int wm = wid & 3, wn = wid >> 2;
    const int bx = blockIdx.x, nb0 = blockIdx.y * 64;

    const bool routed = (bx < 96);
    int e = 0, sdx = 0, bm;
    const float* wB;
    if (routed) { e = bx / 3; bm = bx % 3; wB = wgu + (size_t)e * D * F2; }
    else        { int q = bx - 96; sdx = q >> 4; bm = q & 15; wB = wgu_sh + (size_t)sdx * D * F2; }
    const int m0 = bm * 128;

    if (tid < 128) stok[tid] = routed ? tok_sorted[e * C + m0 + tid] : (m0 + tid);

    const uint32_t sA_base = (uint32_t)(uintptr_t)sA;
    v8f accg[2][2] = {{{0}}}, accu[2][2] = {{{0}}};

    for (int k0 = 0; k0 < D; k0 += 32) {
        __syncthreads();
        if (k0 + 32 < D)   // hint next weight rows into cache (global_prefetch_b8)
            __builtin_prefetch(wB + (size_t)(k0 + 32 + (tid >> 3)) * F2 + nb0, 0, 1);
        // A tile: 128x32 halves = 2048 dwords; async copy does gather + LDS scatter
        #pragma unroll
        for (int it = 0; it < 8; ++it) {
            const int q = it * 256 + tid;
            const int mr = q >> 4, j = q & 15;
            const int t = stok[mr];
            async_copy_b32(sA_base + 4u * a_dword_idx(mr, j),
                           xh + (size_t)t * D + k0 + 2 * j);
        }
        // B tiles: 32x64 fp32 per half; pack 2 k-rows -> one f16x2 dword per store
        #pragma unroll
        for (int it = 0; it < 4; ++it) {
            const int q = it * 256 + tid;           // 1024 (k-pair, n) entries
            const int kp = q >> 6, n = q & 63;
            const int k = kp * 2;
            const float* src = wB + (size_t)(k0 + k) * F2 + nb0 + n;
            const float g0 = src[0],  u0 = src[F];
            const float g1 = src[F2], u1 = src[F2 + F];
            const int idx = b_dword_idx(k, n);
            ((uint32_t*)sBg)[idx] = __builtin_bit_cast(uint32_t, __builtin_amdgcn_cvt_pkrtz(g0, g1));
            ((uint32_t*)sBu)[idx] = __builtin_bit_cast(uint32_t, __builtin_amdgcn_cvt_pkrtz(u0, u1));
        }
        wait_async0();
        __syncthreads();
        v16h a[2], bg[2], bu[2];
        #pragma unroll
        for (int i = 0; i < 2; ++i) a[i]  = ((const v16h*)sA )[(wm * 2 + i) * 32 + lane];
        #pragma unroll
        for (int j = 0; j < 2; ++j) { bg[j] = ((const v16h*)sBg)[(wn * 2 + j) * 32 + lane];
                                      bu[j] = ((const v16h*)sBu)[(wn * 2 + j) * 32 + lane]; }
        #pragma unroll
        for (int i = 0; i < 2; ++i)
            #pragma unroll
            for (int j = 0; j < 2; ++j) {
                accg[i][j] = __builtin_amdgcn_wmma_f32_16x16x32_f16(
                    false, a[i], false, bg[j], (short)0, accg[i][j], false, false);
                accu[i][j] = __builtin_amdgcn_wmma_f32_16x16x32_f16(
                    false, a[i], false, bu[j], (short)0, accu[i][j], false, false);
            }
    }

    // epilogue: act = silu(g) * u
    const int hh = lane >> 4, nn = lane & 15;
    #pragma unroll
    for (int i = 0; i < 2; ++i)
        #pragma unroll
        for (int j = 0; j < 2; ++j)
            #pragma unroll
            for (int r = 0; r < 8; ++r) {
                const int row = m0 + wm * 32 + i * 16 + r + 8 * hh;
                const int col = nb0 + wn * 32 + j * 16 + nn;
                const float g = accg[i][j][r], u = accu[i][j][r];
                const float av = (g / (1.0f + __expf(-g))) * u;
                const size_t dst = routed
                    ? ((size_t)(e * C + row) * F + col)
                    : (ACT_SHARED_OFF_H + (size_t)row * F2 + (size_t)sdx * F + col);
                act[dst] = (_Float16)av;
            }
}

// ============================================================
// 4) down grouped GEMM: routed -> weighted f16 per sorted slot;
//    shared (both experts concatenated along K=2F) -> y (fp32, single writer)
// ============================================================
__global__ __launch_bounds__(256)
void down_kernel(const _Float16* __restrict__ act,
                 const float* __restrict__ wdn,     // [E, F, D]
                 const float* __restrict__ wdn_sh,  // [S, F, D] == [2F, D]
                 const float* __restrict__ wpair,
                 _Float16*    __restrict__ down_pairs, // [E*C, D] f16 (weighted)
                 float*       __restrict__ y)          // [T, D]
{
    __shared__ _Float16 sA[128 * 32];
    __shared__ _Float16 sB[ 32 * 64];
    __shared__ float sScale[128];

    const int tid = threadIdx.x, lane = tid & 31, wid = tid >> 5;
    const int wm = wid & 3, wn = wid >> 2;
    const int bx = blockIdx.x, nb0 = blockIdx.y * 64;

    const bool routed = (bx < 96);
    int e = 0, bm, kmax, rstride;
    const float* wB;
    const _Float16* aBase;
    if (routed) {
        e = bx / 3; bm = bx % 3; kmax = F; rstride = F;
        wB = wdn + (size_t)e * F * D;
        aBase = act + (size_t)(e * C + bm * 128) * F;
    } else {
        bm = bx - 96; kmax = F2; rstride = F2;
        wB = wdn_sh;
        aBase = act + ACT_SHARED_OFF_H + (size_t)(bm * 128) * F2;
    }
    if (tid < 128) sScale[tid] = routed ? wpair[e * C + bm * 128 + tid] : 1.0f;

    const uint32_t sA_base = (uint32_t)(uintptr_t)sA;
    v8f acc[2][2] = {{{0}}};

    for (int k0 = 0; k0 < kmax; k0 += 32) {
        __syncthreads();
        if (k0 + 32 < kmax)
            __builtin_prefetch(wB + (size_t)(k0 + 32 + (tid >> 3)) * D + nb0, 0, 1);
        // A tile (f16 activations): async copy straight into fragment-major LDS
        #pragma unroll
        for (int it = 0; it < 8; ++it) {
            const int q = it * 256 + tid;
            const int mr = q >> 4, j = q & 15;
            async_copy_b32(sA_base + 4u * a_dword_idx(mr, j),
                           aBase + (size_t)mr * rstride + k0 + 2 * j);
        }
        // B tile: 32x64 fp32; pack 2 k-rows into one f16x2 dword
        #pragma unroll
        for (int it = 0; it < 4; ++it) {
            const int q = it * 256 + tid;
            const int kp = q >> 6, n = q & 63;
            const int k = kp * 2;
            const float* src = wB + (size_t)(k0 + k) * D + nb0 + n;
            const float b0 = src[0], b1 = src[D];
            ((uint32_t*)sB)[b_dword_idx(k, n)] =
                __builtin_bit_cast(uint32_t, __builtin_amdgcn_cvt_pkrtz(b0, b1));
        }
        wait_async0();
        __syncthreads();
        v16h a[2], b[2];
        #pragma unroll
        for (int i = 0; i < 2; ++i) a[i] = ((const v16h*)sA)[(wm * 2 + i) * 32 + lane];
        #pragma unroll
        for (int j = 0; j < 2; ++j) b[j] = ((const v16h*)sB)[(wn * 2 + j) * 32 + lane];
        #pragma unroll
        for (int i = 0; i < 2; ++i)
            #pragma unroll
            for (int j = 0; j < 2; ++j)
                acc[i][j] = __builtin_amdgcn_wmma_f32_16x16x32_f16(
                    false, a[i], false, b[j], (short)0, acc[i][j], false, false);
    }

    const int hh = lane >> 4, nn = lane & 15;
    #pragma unroll
    for (int i = 0; i < 2; ++i)
        #pragma unroll
        for (int j = 0; j < 2; ++j)
            #pragma unroll
            for (int r = 0; r < 8; ++r) {
                const int rloc = wm * 32 + i * 16 + r + 8 * hh;   // 0..127
                const int col  = nb0 + wn * 32 + j * 16 + nn;
                if (routed) {
                    const float v = acc[i][j][r] * sScale[rloc];
                    down_pairs[(size_t)(e * C + bm * 128 + rloc) * D + col] = (_Float16)v;
                } else {
                    y[(size_t)(bm * 128 + rloc) * D + col] = acc[i][j][r];
                }
            }
}

// ============================================================
// 5) combine: y[t] += sum_k down_pairs[pos[t*6+k]]  (no atomics)
// ============================================================
__global__ __launch_bounds__(256)
void combine_kernel(const int* __restrict__ pos,
                    const _Float16* __restrict__ down_pairs,
                    float* __restrict__ y)
{
    const int t = blockIdx.x;
    size_t s[TOPK];
    #pragma unroll
    for (int k = 0; k < TOPK; ++k) s[k] = (size_t)pos[t * TOPK + k] * D;
    const size_t yb = (size_t)t * D;
    for (int d = threadIdx.x; d < D; d += 256) {
        float acc = y[yb + d];
        #pragma unroll
        for (int k = 0; k < TOPK; ++k) acc += (float)down_pairs[s[k] + d];
        y[yb + d] = acc;
    }
}

// ============================================================
extern "C" void kernel_launch(void* const* d_in, const int* in_sizes, int n_in,
                              void* d_out, int out_size, void* d_ws, size_t ws_size,
                              hipStream_t stream)
{
    const float* x       = (const float*)d_in[0];
    const float* weights = (const float*)d_in[1];
    const float* wgu     = (const float*)d_in[2];
    const float* wdn     = (const float*)d_in[3];
    const float* wgu_sh  = (const float*)d_in[4];
    const float* wdn_sh  = (const float*)d_in[5];
    const int*   indices = (const int*)d_in[6];
    float* y = (float*)d_out;

    char* ws = (char*)d_ws;
    int*       tok = (int*)      (ws + OFF_TOK);
    float*     wp  = (float*)    (ws + OFF_WPAIR);
    int*       pos = (int*)      (ws + OFF_POS);
    _Float16*  xh  = (_Float16*) (ws + OFF_XH);
    _Float16*  act = (_Float16*) (ws + OFF_ACT);
    _Float16*  dp  = (_Float16*) (ws + OFF_DOWN);

    dispatch_kernel  <<<E, 32, 0, stream>>>(indices, weights, tok, wp, pos);
    convert_x_kernel <<<(T * D / 4) / 256, 256, 0, stream>>>(x, xh);
    // 96 routed M-blocks + 2 shared experts * 16 M-blocks = 128; N = F/64 = 22
    gateup_kernel    <<<dim3(128, F / 64), 256, 0, stream>>>(xh, wgu, wgu_sh, tok, act);
    // 96 routed M-blocks + 16 shared M-blocks = 112; N = D/64 = 32
    down_kernel      <<<dim3(112, D / 64), 256, 0, stream>>>(act, wdn, wdn_sh, wp, dp, y);
    combine_kernel   <<<T, 256, 0, stream>>>(pos, dp, y);
}
